// MultiHeadAttention_56066503082210
// MI455X (gfx1250) — compile-verified, matrix operand order checked by reading, not run on hardware
//
#include <hip/hip_runtime.h>
#include <hip/hip_bf16.h>

typedef __attribute__((ext_vector_type(16))) __bf16 v16bf;
typedef __attribute__((ext_vector_type(4)))  __bf16 v4bf;
typedef __attribute__((ext_vector_type(8)))  float  v8f;
typedef __attribute__((ext_vector_type(4)))  int    v4i;

#define DIMSZ 1024
#define NTOK  1024
#define NHEAD 16
#define HDIM  64

#if defined(__has_builtin)
#if __has_builtin(__builtin_amdgcn_global_load_async_to_lds_b128)
#define HAVE_ASYNC_LDS 1
#endif
#endif
#ifndef HAVE_ASYNC_LDS
#define HAVE_ASYNC_LDS 0
#endif

union Frag32 { v16bf v; uint4 u[2]; };

// 16x32 bf16 WMMA operand fragment: per-lane two contiguous 16B chunks at
// p and p+16 elements (caller applies the lane's row/half offsets).
__device__ __forceinline__ v16bf ld_frag(const __bf16* p) {
  Frag32 f;
  f.u[0] = *(const uint4*)(p);
  f.u[1] = *(const uint4*)(p + 16);
  return f.v;
}

__device__ __forceinline__ v8f wmma_bf16(v16bf a, v16bf b, v8f c) {
  return __builtin_amdgcn_wmma_f32_16x16x32_bf16(false, a, false, b, (short)0, c,
                                                 false, false);
}

__device__ __forceinline__ v4bf pack_bf4(float4 f) {
  v4bf r;
  r[0] = (__bf16)f.x; r[1] = (__bf16)f.y; r[2] = (__bf16)f.z; r[3] = (__bf16)f.w;
  return r;
}

// 16-byte global -> LDS copy: async (CDNA5 GLOBAL_LOAD_ASYNC_TO_LDS_B128,
// ASYNCcnt-tracked) when the builtin exists, else via VGPRs.
// Builtin signature (from hipcc diagnostic): (v4i as1*, v4i as3*, imm, imm).
__device__ __forceinline__ void cp16_g2l(const __bf16* gsrc, __bf16* ldst) {
#if HAVE_ASYNC_LDS
  typedef __attribute__((address_space(1))) v4i* gp_t;
  typedef __attribute__((address_space(3))) v4i* lp_t;
  __builtin_amdgcn_global_load_async_to_lds_b128((gp_t)gsrc, (lp_t)ldst, 0, 0);
#else
  *(uint4*)ldst = *(const uint4*)gsrc;
#endif
}

__device__ __forceinline__ void cp16_wait() {
#if HAVE_ASYNC_LDS
#if __has_builtin(__builtin_amdgcn_s_wait_asynccnt)
  __builtin_amdgcn_s_wait_asynccnt(0);
#else
  asm volatile("s_wait_asynccnt 0" ::: "memory");
#endif
#endif
}

// ---------------------------------------------------------------------------
// Fused GEMM: C = A @ W^T over K=1024, 64x64 tile per block (4 waves),
// double-buffered LDS, K-chunk 64, fp32->bf16 conversion in staging regs.
// mode 0/1: q/k-proj -> RMSNorm + RoPE -> bf16 [B,H,N,hd]
// mode 2:   v-proj   -> bf16 transposed [B,H,hd,N]
// mode 3:   out-proj -> fp32 [B,N,DIM]
// ---------------------------------------------------------------------------
__global__ __launch_bounds__(128)
void fused_gemm_kernel(const float* __restrict__ A,
                       const float* __restrict__ W0,
                       const float* __restrict__ W1,
                       const float* __restrict__ W2,
                       const float* __restrict__ q_gamma,
                       const float* __restrict__ k_gamma,
                       const float* __restrict__ cosT,
                       const float* __restrict__ sinT,
                       const int*   __restrict__ rope_idx,
                       unsigned short* __restrict__ qb_u,
                       unsigned short* __restrict__ kb_u,
                       unsigned short* __restrict__ vt_u,
                       float* __restrict__ outf,
                       int proj)
{
  __shared__ __align__(16) __bf16 As[2][64][72];  // 144B row stride (16B mult)
  __shared__ __align__(16) __bf16 Bs[2][64][72];
  __shared__ float Cs[64][65];                    // +1 pad: conflict-free

  const int mode = (proj < 0) ? (int)blockIdx.z : proj;
  const float* W = (mode == 1) ? W1 : (mode == 2) ? W2 : W0;
  const int tid  = threadIdx.x;
  const int lane = tid & 31;
  const int wave = tid >> 5;
  const int r16  = lane & 15;
  const int hl   = lane >> 4;
  const int mrow = blockIdx.y * 64;
  const int ncol = blockIdx.x * 64;
  const int wrow = (wave >> 1) * 32;
  const int wcol = (wave & 1) * 32;

  v8f c00, c01, c10, c11;
  #pragma unroll
  for (int i = 0; i < 8; ++i) { c00[i]=0.f; c01[i]=0.f; c10[i]=0.f; c11[i]=0.f; }

  v4bf sa[8], sb[8];                              // bf16 staging registers

  auto stage_load = [&](int kc) {                 // global fp32 -> bf16 regs
    #pragma unroll
    for (int t = 0; t < 8; ++t) {
      int i = tid + t * 128;                      // [0, 1024)
      int row = i >> 4, c4 = (i & 15) << 2;
      sa[t] = pack_bf4(*(const float4*)(A + (size_t)(mrow + row) * DIMSZ + kc + c4));
      sb[t] = pack_bf4(*(const float4*)(W + (size_t)(ncol + row) * DIMSZ + kc + c4));
    }
  };
  auto stage_store = [&](int buf) {               // regs -> LDS tile
    #pragma unroll
    for (int t = 0; t < 8; ++t) {
      int i = tid + t * 128;
      int row = i >> 4, c4 = (i & 15) << 2;
      *(v4bf*)&As[buf][row][c4] = sa[t];
      *(v4bf*)&Bs[buf][row][c4] = sb[t];
    }
  };

  stage_load(0);
  stage_store(0);
  __syncthreads();

  for (int it = 0; it < 16; ++it) {
    int cur = it & 1;
    if (it < 15) stage_load((it + 1) * 64);       // overlap with WMMAs below
    #pragma unroll
    for (int kk = 0; kk < 64; kk += 32) {
      v16bf a0 = ld_frag(&As[cur][wrow      + r16][kk + hl * 8]);
      v16bf a1 = ld_frag(&As[cur][wrow + 16 + r16][kk + hl * 8]);
      v16bf b0 = ld_frag(&Bs[cur][wcol      + r16][kk + hl * 8]);
      v16bf b1 = ld_frag(&Bs[cur][wcol + 16 + r16][kk + hl * 8]);
      c00 = wmma_bf16(a0, b0, c00);
      c01 = wmma_bf16(a0, b1, c01);
      c10 = wmma_bf16(a1, b0, c10);
      c11 = wmma_bf16(a1, b1, c11);
    }
    if (it < 15) stage_store(1 - cur);
    __syncthreads();
  }

  // C layout: VGPR r -> row (r + hl*8) of the 16-block; lane%16 -> col
  #pragma unroll
  for (int r = 0; r < 8; ++r) {
    Cs[wrow +      hl*8 + r][wcol +      r16] = c00[r];
    Cs[wrow +      hl*8 + r][wcol + 16 + r16] = c01[r];
    Cs[wrow + 16 + hl*8 + r][wcol +      r16] = c10[r];
    Cs[wrow + 16 + hl*8 + r][wcol + 16 + r16] = c11[r];
  }
  __syncthreads();

  if (mode == 3) {                                // plain fp32 store (Wo proj)
    for (int i = tid; i < 64 * 64; i += 128) {
      int row = i >> 6, col = i & 63;
      outf[(size_t)(mrow + row) * DIMSZ + ncol + col] = Cs[row][col];
    }
    return;
  }

  const int b   = mrow >> 10;                     // 64 | 1024: batch constant
  const int nt0 = mrow & 1023;
  const int h   = blockIdx.x;                     // N-tile == head

  if (mode == 2) {                                // V: store transposed bf16
    __bf16* vt = (__bf16*)vt_u;
    for (int i = tid; i < 64 * 64; i += 128) {
      int d = i >> 6, nn = i & 63;                // nn fastest -> coalesced
      vt[(((size_t)b * NHEAD + h) * HDIM + d) * NTOK + nt0 + nn] =
          (__bf16)Cs[nn][d];
    }
    return;
  }

  // Q/K: per-row RMSNorm over head_dim + gathered 2D RoPE, bf16 store
  const float* gam = (mode == 0) ? q_gamma : k_gamma;
  __bf16* outp = (__bf16*)((mode == 0) ? qb_u : kb_u);
  if (tid < 64) {
    int row = tid;
    int ntok = nt0 + row;
    float ss = 0.f;
    #pragma unroll
    for (int j = 0; j < 64; ++j) { float v = Cs[row][j]; ss += v * v; }
    float rs = rsqrtf(ss * (1.0f / 64.0f) + 1e-6f);
    size_t obase = (((size_t)b * NHEAD + h) * NTOK + ntok) * HDIM;
    int idx = rope_idx[ntok];
    if (idx >= 0) {
      const float* cr = cosT + (size_t)idx * HDIM;
      const float* sr = sinT + (size_t)idx * HDIM;
      #pragma unroll
      for (int j = 0; j < 32; ++j) {
        float a  = Cs[row][j]      * rs * gam[j];
        float bq = Cs[row][j + 32] * rs * gam[j + 32];
        outp[obase + j]      = (__bf16)(a * cr[j]       - bq * sr[j]);
        outp[obase + j + 32] = (__bf16)(bq * cr[j + 32] + a  * sr[j + 32]);
      }
    } else {                                      // -1 sentinel: no rotation
      #pragma unroll
      for (int j = 0; j < 64; ++j)
        outp[obase + j] = (__bf16)(Cs[row][j] * rs * gam[j]);
    }
  }
}

// ---------------------------------------------------------------------------
// Flash attention: one (b,h) per blockIdx.y, 64 query rows per block.
// K/V tiles are shared by all 4 waves -> stage them once per block in
// double-buffered LDS (async global->LDS when available), 4x less L2 traffic.
// ---------------------------------------------------------------------------
__global__ __launch_bounds__(128)
void flash_attn_kernel(const unsigned short* __restrict__ qb_u,
                       const unsigned short* __restrict__ kb_u,
                       const unsigned short* __restrict__ vt_u,
                       const unsigned char*  __restrict__ mask,
                       float* __restrict__ attn_out)
{
  __shared__ __align__(16) __bf16 Ks[2][32][72];     // keys x hd, 144B stride
  __shared__ __align__(16) __bf16 Vs[2][64][40];     // hd x keys, 80B stride
  __shared__ __align__(16) __bf16 Plds[4][16][32];   // per-wave P buffer
  const __bf16* qb = (const __bf16*)qb_u;
  const __bf16* kb = (const __bf16*)kb_u;
  const __bf16* vt = (const __bf16*)vt_u;

  const int tid  = threadIdx.x;
  const int lane = tid & 31;
  const int wave = tid >> 5;
  const int r16  = lane & 15;
  const int hl   = lane >> 4;
  const int bh   = blockIdx.y;                    // b*16 + h
  const int b    = bh >> 4;
  const int h    = bh & 15;
  const int qbase = blockIdx.x * 64 + wave * 16;

  auto stage_kv = [&](int buf, int kt) {          // 4KB K + 4KB V per tile
    #pragma unroll
    for (int t = 0; t < 2; ++t) {
      int i = tid + t * 128;                      // [0, 256)
      int krow = i >> 3, kc = (i & 7) << 3;
      cp16_g2l(kb + ((size_t)bh * NTOK + kt + krow) * HDIM + kc,
               &Ks[buf][krow][kc]);
      int vrow = i >> 2, vc = (i & 3) << 3;
      cp16_g2l(vt + ((size_t)bh * HDIM + vrow) * NTOK + kt + vc,
               &Vs[buf][vrow][vc]);
    }
  };

  // Q operand: 16x64 -> two 16x32 A fragments, resident for whole kernel
  const __bf16* qrow = qb + ((size_t)bh * NTOK + qbase + r16) * HDIM + hl * 8;
  v16bf aq0 = ld_frag(qrow);
  v16bf aq1 = ld_frag(qrow + 32);

  v8f o0, o1, o2, o3;
  float m[8], l[8];
  #pragma unroll
  for (int i = 0; i < 8; ++i) {
    o0[i]=0.f; o1[i]=0.f; o2[i]=0.f; o3[i]=0.f; m[i]=-1e30f; l[i]=0.f;
  }

  const float SCALE = 0.125f;                     // hd^-0.5
  const float CAP = 50.0f, INVCAP = 0.02f;

  stage_kv(0, 0);
  cp16_wait();
  __syncthreads();

  for (int it = 0; it < 32; ++it) {
    const int cur = it & 1;
    const int kt  = it * 32;
    if (it < 31) stage_kv(1 - cur, kt + 32);      // prefetch next tile

    v8f slo, shi;
    #pragma unroll
    for (int i = 0; i < 8; ++i) { slo[i]=0.f; shi[i]=0.f; }
    slo = wmma_bf16(aq0, ld_frag(&Ks[cur][r16     ][     hl * 8]), slo);
    slo = wmma_bf16(aq1, ld_frag(&Ks[cur][r16     ][32 + hl * 8]), slo);
    shi = wmma_bf16(aq0, ld_frag(&Ks[cur][16 + r16][     hl * 8]), shi);
    shi = wmma_bf16(aq1, ld_frag(&Ks[cur][16 + r16][32 + hl * 8]), shi);

    #pragma unroll
    for (int r = 0; r < 8; ++r) {
      int row = qbase + hl * 8 + r;
      const unsigned char* mp = mask + (size_t)row * NTOK + kt;
      float tlo = slo[r] * SCALE;
      float thi = shi[r] * SCALE;
      tlo = CAP * tanhf(tlo * INVCAP);
      thi = CAP * tanhf(thi * INVCAP);
      if (!mp[r16])      tlo = -3.0e38f;          // finite -inf surrogate
      if (!mp[16 + r16]) thi = -3.0e38f;
      float rm = fmaxf(tlo, thi);
      #pragma unroll
      for (int off = 1; off < 16; off <<= 1)
        rm = fmaxf(rm, __shfl_xor(rm, off, 32));
      float mnew = fmaxf(m[r], rm);
      float corr = __expf(m[r] - mnew);
      float plo  = __expf(tlo - mnew);
      float phi  = __expf(thi - mnew);
      float rsum = plo + phi;
      #pragma unroll
      for (int off = 1; off < 16; off <<= 1)
        rsum += __shfl_xor(rsum, off, 32);
      l[r] = l[r] * corr + rsum;
      m[r] = mnew;
      o0[r] *= corr; o1[r] *= corr; o2[r] *= corr; o3[r] *= corr;
      Plds[wave][hl * 8 + r][r16]      = (__bf16)plo;  // C-layout -> LDS
      Plds[wave][hl * 8 + r][16 + r16] = (__bf16)phi;
    }
    asm volatile("s_wait_dscnt 0" ::: "memory");       // per-wave LDS RAW
    v16bf ap = ld_frag(&Plds[wave][r16][hl * 8]);      // reload in A layout

    o0 = wmma_bf16(ap, ld_frag(&Vs[cur][r16     ][hl * 8]), o0);
    o1 = wmma_bf16(ap, ld_frag(&Vs[cur][16 + r16][hl * 8]), o1);
    o2 = wmma_bf16(ap, ld_frag(&Vs[cur][32 + r16][hl * 8]), o2);
    o3 = wmma_bf16(ap, ld_frag(&Vs[cur][48 + r16][hl * 8]), o3);

    cp16_wait();                                  // async stores landed
    __syncthreads();
  }

  #pragma unroll
  for (int r = 0; r < 8; ++r) {
    int row = qbase + hl * 8 + r;
    float inv = 1.0f / l[r];
    size_t base = ((size_t)b * NTOK + row) * DIMSZ + h * HDIM + r16;
    attn_out[base]      = o0[r] * inv;
    attn_out[base + 16] = o1[r] * inv;
    attn_out[base + 32] = o2[r] * inv;
    attn_out[base + 48] = o3[r] * inv;
  }
}

extern "C" void kernel_launch(void* const* d_in, const int* in_sizes, int n_in,
                              void* d_out, int out_size, void* d_ws, size_t ws_size,
                              hipStream_t stream) {
  (void)in_sizes; (void)n_in; (void)out_size; (void)ws_size;
  const float* x    = (const float*)d_in[0];
  const float* Wq   = (const float*)d_in[1];
  const float* Wk   = (const float*)d_in[2];
  const float* Wv   = (const float*)d_in[3];
  const float* Wo   = (const float*)d_in[4];
  const float* qg   = (const float*)d_in[5];
  const float* kg   = (const float*)d_in[6];
  const float* cosT = (const float*)d_in[7];
  const float* sinT = (const float*)d_in[8];
  const int*   ridx = (const int*)d_in[9];
  const unsigned char* mask = (const unsigned char*)d_in[10];

  char* ws = (char*)d_ws;
  unsigned short* qb  = (unsigned short*)(ws);                            // 16MB bf16 [B,H,N,hd]
  unsigned short* kbp = (unsigned short*)(ws + (size_t)16 * 1024 * 1024); // 16MB bf16
  unsigned short* vt  = (unsigned short*)(ws + (size_t)32 * 1024 * 1024); // 16MB bf16 [B,H,hd,N]
  float* attnws       = (float*)(ws + (size_t)48 * 1024 * 1024);          // 32MB fp32 [B*N, DIM]

  dim3 blk(128);
  // 1) fused QKV projections + QKNorm + RoPE + bf16 pack (blockIdx.z = q/k/v)
  fused_gemm_kernel<<<dim3(16, 128, 3), blk, 0, stream>>>(
      x, Wq, Wk, Wv, qg, kg, cosT, sinT, ridx, qb, kbp, vt, nullptr, -1);
  // 2) flash attention with soft cap + mask + fp32 online softmax
  flash_attn_kernel<<<dim3(16, 128), blk, 0, stream>>>(qb, kbp, vt, mask, attnws);
  // 3) output projection -> fp32 result
  fused_gemm_kernel<<<dim3(16, 128, 1), blk, 0, stream>>>(
      attnws, Wo, Wo, Wo, qg, kg, cosT, sinT, ridx, qb, kbp, vt,
      (float*)d_out, 3);
}
